// RelativeIlluminationField_20100446945928
// MI455X (gfx1250) — matrix operand order repaired and checked by reading, not compiled, not run on hardware
//
#include <hip/hip_runtime.h>
#include <hip/hip_bf16.h>

typedef __attribute__((ext_vector_type(16))) _Float16 v16h;
typedef __attribute__((ext_vector_type(8)))  float    v8f;

#define NGP_LEVELS 8
#define NGP_TABLE  131072           // 2^17 -> mod is AND

// floor(16 * (64^(1/7))^l), l=0..7  (matches numpy float64 evaluation)
__constant__ float kScale[NGP_LEVELS] = {
    16.f, 28.f, 52.f, 95.f, 172.f, 312.f, 565.f, 1024.f};

__device__ __forceinline__ unsigned asu(float f) { return __float_as_uint(f); }
__device__ __forceinline__ float    asf(unsigned u) { return __uint_as_float(u); }

__device__ __forceinline__ unsigned packh2(float a, float b) {
    union { _Float16 h[2]; unsigned u; } c;
    c.h[0] = (_Float16)a; c.h[1] = (_Float16)b;
    return c.u;
}

// Half-wave exchange producing BOTH combinations in one step:
//   X = { B[16..31] -> lanes 0-15 , A[16..31] stays in lanes 16-31 }
//   Y = { B[0..15]  stays in lanes 0-15 , A[0..15] -> lanes 16-31 }
__device__ __forceinline__ void halfswap(unsigned a, unsigned b, int laneHi,
                                         unsigned& X, unsigned& Y) {
#if __has_builtin(__builtin_amdgcn_permlane16_swap)
    typedef __attribute__((ext_vector_type(2))) unsigned u2v;
    u2v r = __builtin_amdgcn_permlane16_swap(a, b, false, false);
    X = r.x; Y = r.y; (void)laneHi;
#else
    // SWAPX16 via ds_swizzle immediate (xor=0x10, and=0x1f): lane <- lane^16
    const unsigned as_ = __builtin_amdgcn_ds_swizzle(a, 0x401f);
    const unsigned bs_ = __builtin_amdgcn_ds_swizzle(b, 0x401f);
    X = laneHi ? a : bs_;
    Y = laneHi ? as_ : b;
#endif
}

__global__ __launch_bounds__(256) void
ngp_fused_kernel(const float* __restrict__ positions,
                 const float* __restrict__ c2ws,
                 const float* __restrict__ aabb,
                 const float* __restrict__ table,
                 const float* __restrict__ w1,  const float* __restrict__ b1,
                 const float* __restrict__ w2,  const float* __restrict__ b2,
                 const float* __restrict__ hw1, const float* __restrict__ hb1,
                 const float* __restrict__ hw2, const float* __restrict__ hb2,
                 float* __restrict__ out, int n_points, int S)
{
    const int lane   = threadIdx.x & 31;
    const int laneHi = (lane >> 4) & 1;          // 0: lanes 0-15, 1: lanes 16-31
    const int lm     = lane & 15;

    // ---------------- loop-invariant weight fragments (WMMA A operand, this code's contiguous K order) ----
    // A 16x32 f16: lanes0-15 M=lane, element h: K=h;
    //              lanes16-31 same M, K offset +16.
    v16h a1t0, a1t1, a2f, a3f[4];
#pragma unroll
    for (int h = 0; h < 16; ++h) {
        const int r = h >> 1, pos = h & 1;
        const int k = 16 * laneHi + 2 * r + pos;
        a2f[h] = (_Float16)w2[k * 16 + lm];               // K=32 exact
        if (k < 16) {
            a1t0[h]   = (_Float16)w1[k * 32 + lm];
            a1t1[h]   = (_Float16)w1[k * 32 + 16 + lm];
            a3f[0][h] = (_Float16)hw1[k * 64 + lm];
            a3f[1][h] = (_Float16)hw1[k * 64 + 16 + lm];
            a3f[2][h] = (_Float16)hw1[k * 64 + 32 + lm];
            a3f[3][h] = (_Float16)hw1[k * 64 + 48 + lm];
        } else {                                           // K padding
            a1t0[h] = (_Float16)0.f; a1t1[h] = (_Float16)0.f;
            a3f[0][h] = (_Float16)0.f; a3f[1][h] = (_Float16)0.f;
            a3f[2][h] = (_Float16)0.f; a3f[3][h] = (_Float16)0.f;
        }
    }
    // Bias rows per C/D layout: VGPR r holds m = r + 8*laneHi (+16*tile)
    float b1v[2][8], b2v[8], hb1v[4][8], w4v[4][8];
#pragma unroll
    for (int r = 0; r < 8; ++r) {
        const int m = r + 8 * laneHi;
        b1v[0][r] = b1[m];        b1v[1][r] = b1[16 + m];
        b2v[r]    = b2[m];
#pragma unroll
        for (int t = 0; t < 4; ++t) {
            hb1v[t][r] = hb1[16 * t + m];
            w4v[t][r]  = hw2[16 * t + m];
        }
    }
    const float hb2s = hb2[0];
    const float a0x = aabb[0], a0y = aabb[1], a0z = aabb[2];
    const float r1x = 1.f / (aabb[3] - a0x), r1y = 1.f / (aabb[4] - a0y),
                r1z = 1.f / (aabb[5] - a0z);

    const int warpId   = (blockIdx.x * blockDim.x + threadIdx.x) >> 5;
    const int warpStep = (gridDim.x * blockDim.x) >> 5;
    const int chunks   = (n_points + 31) >> 5;

    for (int chunk = warpId; chunk < chunks; chunk += warpStep) {
        const int base = chunk << 5;
        int p = base + lane; if (p >= n_points) p = n_points - 1;

        // prefetch next chunk's positions into cache
        {
            long long np_ = (long long)(base + warpStep * 32 + lane);
            if (np_ < n_points) __builtin_prefetch(&positions[3 * np_], 0, 1);
        }

        // ---------------- per-lane: transform + hash-grid encode ---------------
        const float P0 = positions[3 * p + 0], P1 = positions[3 * p + 1],
                    P2 = positions[3 * p + 2];
        const float* M = c2ws + (size_t)(p / S) * 12;
        const float q0 = P0 - M[3], q1 = P1 - M[7], q2 = P2 - M[11];
        // pos_cam[i] = sum_j rot[j][i]*(P[j]-t[j]),  rot[j][i] = M[j*4+i]
        float x = M[0] * q0 + M[4] * q1 + M[8]  * q2;
        float y = M[1] * q0 + M[5] * q1 + M[9]  * q2;
        float z = M[2] * q0 + M[6] * q1 + M[10] * q2;
        x = (x - a0x) * r1x; y = (y - a0y) * r1y; z = (z - a0z) * r1z;
        const bool inb = (x > 0.f) & (x < 1.f) & (y > 0.f) & (y < 1.f) &
                         (z > 0.f) & (z < 1.f);
        const float fsel = inb ? 1.f : 0.f;
        x *= fsel; y *= fsel; z *= fsel;

        float enc[16];
#pragma unroll
        for (int l = 0; l < NGP_LEVELS; ++l) {
            const float s = kScale[l];
            const float sx = x * s, sy = y * s, sz = z * s;
            const float fx = floorf(sx), fy = floorf(sy), fz = floorf(sz);
            const float wx = sx - fx, wy = sy - fy, wz = sz - fz;
            const unsigned ix = (unsigned)(int)fx, iy = (unsigned)(int)fy,
                           iz = (unsigned)(int)fz;
            const unsigned hy0 = iy * 2654435761u, hy1 = (iy + 1u) * 2654435761u;
            const unsigned hz0 = iz * 805459861u,  hz1 = (iz + 1u) * 805459861u;
            float e0 = 0.f, e1 = 0.f;
#pragma unroll
            for (int c = 0; c < 8; ++c) {
                const unsigned cx = (c >> 2) & 1, cy = (c >> 1) & 1, cz = c & 1;
                const unsigned h = (ix + cx) ^ (cy ? hy1 : hy0) ^ (cz ? hz1 : hz0);
                const unsigned row = (h & (NGP_TABLE - 1)) + (unsigned)l * NGP_TABLE;
                const float2 f = ((const float2*)table)[row];      // 8B gather (L2)
                const float cw = (cx ? wx : 1.f - wx) * (cy ? wy : 1.f - wy) *
                                 (cz ? wz : 1.f - wz);
                e0 = fmaf(cw, f.x, e0); e1 = fmaf(cw, f.y, e1);
            }
            enc[2 * l] = e0; enc[2 * l + 1] = e1;
        }

        // pack features to f16 pairs
        unsigned encp[8];
#pragma unroll
        for (int r = 0; r < 8; ++r) encp[r] = packh2(enc[2 * r], enc[2 * r + 1]);

        // B1 fragments for BOTH point-tiles in one pass of half-swaps:
        // tile0 lane n needs point n's feats (lanes16-31 = K-pad zeros);
        // tile1 lane n needs point 16+n's feats.
        union Pk { unsigned u[8]; v16h v; } b1f0, b1f1;
#pragma unroll
        for (int r = 0; r < 8; ++r)
            halfswap(0u, encp[r], laneHi, b1f1.u[r], b1f0.u[r]);

        unsigned fsX, fsY;
        halfswap(asu(fsel), asu(fsel), laneHi, fsX, fsY);
        const float fselx = asf(fsX);        // lanes0-15: partner point's fsel

        // ---------------- MLP via WMMA, two 16-point tiles per wave -------------
#pragma unroll
        for (int nt = 0; nt < 2; ++nt) {
            const v16h bB1 = nt ? b1f1.v : b1f0.v;

            v8f acc;
#pragma unroll
            for (int r = 0; r < 8; ++r) acc[r] = b1v[0][r];
            v8f d1a = __builtin_amdgcn_wmma_f32_16x16x32_f16(
                false, a1t0, false, bB1, (short)0, acc, false, false);
#pragma unroll
            for (int r = 0; r < 8; ++r) acc[r] = b1v[1][r];
            v8f d1b = __builtin_amdgcn_wmma_f32_16x16x32_f16(
                false, a1t1, false, bB1, (short)0, acc, false, false);

            float r1a[8], r1b[8];
#pragma unroll
            for (int r = 0; r < 8; ++r) {
                r1a[r] = fmaxf(d1a[r], 0.f); r1b[r] = fmaxf(d1b[r], 0.f);
            }
            // B2 (32 feats x 16 pts): elem r = {r1a.lo, r1b.lo},
            //                         elem r+8 = {r1a.hi, r1b.hi}
            v16h bB2;
#pragma unroll
            for (int r = 0; r < 8; ++r) {
                unsigned X, Y;
                halfswap(asu(r1b[r]), asu(r1a[r]), laneHi, X, Y);
                bB2[r]     = (_Float16)asf(Y);
                bB2[r + 8] = (_Float16)asf(X);
            }
#pragma unroll
            for (int r = 0; r < 8; ++r) acc[r] = b2v[r];
            v8f h2 = __builtin_amdgcn_wmma_f32_16x16x32_f16(
                false, a2f, false, bB2, (short)0, acc, false, false);   // no relu

            // B3 (K=16 real, lanes16-31 zero): elem r = {h2.lo,0}, r+8 = {h2.hi,0}
            v16h bB3;
#pragma unroll
            for (int r = 0; r < 8; ++r) {
                unsigned X, Y;
                halfswap(0u, asu(h2[r]), laneHi, X, Y);
                bB3[r]     = (_Float16)asf(Y);
                bB3[r + 8] = (_Float16)asf(X);
            }

            float partial = 0.f;
#pragma unroll
            for (int t = 0; t < 4; ++t) {
#pragma unroll
                for (int r = 0; r < 8; ++r) acc[r] = hb1v[t][r];
                v8f d3 = __builtin_amdgcn_wmma_f32_16x16x32_f16(
                    false, a3f[t], false, bB3, (short)0, acc, false, false);
#pragma unroll
                for (int r = 0; r < 8; ++r)
                    partial = fmaf(fmaxf(d3[r], 0.f), w4v[t][r], partial);
            }
            unsigned pX, pY;
            halfswap(asu(partial), asu(partial), laneHi, pX, pY);
            const float o = partial + asf(pX) + hb2s;    // valid in lanes 0-15
            const float sel = nt ? fselx : fsel;
            const float res = sel / (1.f + __expf(5.0f - o));   // sel*sigmoid(o-5)

            const int idx = base + nt * 16 + lm;
            if (!laneHi && idx < n_points) out[idx] = res;
        }
    }
}

extern "C" void kernel_launch(void* const* d_in, const int* in_sizes, int n_in,
                              void* d_out, int out_size, void* d_ws, size_t ws_size,
                              hipStream_t stream) {
    const float* positions = (const float*)d_in[0];
    const float* c2ws      = (const float*)d_in[1];
    const float* aabb      = (const float*)d_in[2];
    const float* table     = (const float*)d_in[3];
    const float* w1  = (const float*)d_in[4];
    const float* b1  = (const float*)d_in[5];
    const float* w2  = (const float*)d_in[6];
    const float* b2  = (const float*)d_in[7];
    const float* hw1 = (const float*)d_in[8];
    const float* hb1 = (const float*)d_in[9];
    const float* hw2 = (const float*)d_in[10];
    const float* hb2 = (const float*)d_in[11];
    float* out = (float*)d_out;

    const int n_points = in_sizes[0] / 3;            // R*S
    const int n_rays   = in_sizes[1] / 12;           // R
    const int S        = n_points / n_rays;          // samples per ray

    const int threads = 256;                          // 8 waves / block
    const int chunks  = (n_points + 31) / 32;
    int blocks = (chunks + (threads / 32) - 1) / (threads / 32);
    if (blocks > 4096) blocks = 4096;                 // grid-stride covers rest

    ngp_fused_kernel<<<blocks, threads, 0, stream>>>(
        positions, c2ws, aabb, table, w1, b1, w2, b2, hw1, hb1, hw2, hb2,
        out, n_points, S);
}
